// MIMOWhoGeneralDotProductAttention_22256520528801
// MI455X (gfx1250) — matrix operand order repaired and checked by reading, not compiled, not run on hardware
//
#include <hip/hip_runtime.h>
#include <math.h>

typedef __attribute__((ext_vector_type(16))) _Float16 v16h;
typedef __attribute__((ext_vector_type(8)))  float    v8f;
typedef __attribute__((ext_vector_type(4)))  float    f4;

#define B_    32
#define N_    10
#define QS_   128
#define KS_   256
#define CHW_  524288                       // 512*32*32
#define OUT_ELEMS ((size_t)B_ * N_ * CHW_) // 167772160

// ---------------------------------------------------------------------------
// Kernel 1: one wave (32 lanes) per batch b.
//   query = qu @ W^T + bias      (WMMA f32<-f16, 16x16x32, 16 col tiles x 4 K-steps)
//   attn  = k @ query^T          (WMMA f32<-f16, 1 tile x 8 K-steps)
//   app   = reinsert-diag-zeros(softmax(delete-diag(attn), axis=rows))
// ---------------------------------------------------------------------------
__global__ __launch_bounds__(32) void attn_app_kernel(
    const float* __restrict__ qu, const float* __restrict__ kmat,
    const float* __restrict__ W,  const float* __restrict__ bias,
    float* __restrict__ app_out)
{
    __shared__ float q_lds[16][KS_];   // query tile rows 0..15 (rows >=10 unused)
    __shared__ float attn_lds[16][17];

    const int b    = blockIdx.x;
    const int lane = threadIdx.x;      // 0..31, wave32
    const int mrow = lane & 15;        // M row (A/C) == N col (B/C) per lane
    const int half = lane >> 4;

    const float* quB = qu   + (size_t)b * N_ * QS_;
    const float* kB  = kmat + (size_t)b * N_ * KS_;

    // ---- Step 1: query = qu @ W^T + bias -------------------------------
    // Hoist the 4 A-fragments (K-steps over qu's 128 columns).
    v16h afrag[4];
#pragma unroll
    for (int s = 0; s < 4; ++s) {
#pragma unroll
        for (int v = 0; v < 8; ++v) {
            int k0 = ((v < 4) ? 2 * v : 16 + 2 * (v - 4)) + 8 * half;
            int d  = s * 32 + k0;
            float x0 = (mrow < N_) ? quB[mrow * QS_ + d]     : 0.0f;
            float x1 = (mrow < N_) ? quB[mrow * QS_ + d + 1] : 0.0f;
            afrag[s][2 * v]     = (_Float16)x0;
            afrag[s][2 * v + 1] = (_Float16)x1;
        }
    }

    for (int ct = 0; ct < 16; ++ct) {          // 16 column tiles of 16 -> KS=256
        v8f acc = {};
#pragma unroll
        for (int s = 0; s < 4; ++s) {
            v16h bfrag;
#pragma unroll
            for (int v = 0; v < 8; ++v) {
                int k0 = ((v < 4) ? 2 * v : 16 + 2 * (v - 4)) + 8 * half;
                int d  = s * 32 + k0;
                // B[K=d][N=mrow] = W[ct*16+mrow][d]
                bfrag[2 * v]     = (_Float16)W[(ct * 16 + mrow) * QS_ + d];
                bfrag[2 * v + 1] = (_Float16)W[(ct * 16 + mrow) * QS_ + d + 1];
            }
            acc = __builtin_amdgcn_wmma_f32_16x16x32_f16(
                false, afrag[s], false, bfrag, (short)0, acc, false, false);
        }
        float bb = bias[ct * 16 + mrow];       // bias indexed by output column
#pragma unroll
        for (int r = 0; r < 8; ++r)
            q_lds[r + 8 * half][ct * 16 + mrow] = acc[r] + bb;
    }
    __syncthreads();

    // ---- Step 2: attn = k @ query^T  (K-dim = 256 -> 8 steps) ----------
    v8f atile = {};
#pragma unroll
    for (int s = 0; s < 8; ++s) {
        v16h ka, qb;
#pragma unroll
        for (int v = 0; v < 8; ++v) {
            int k0a = ((v < 4) ? 2 * v : 16 + 2 * (v - 4)) + 8 * half;
            int da  = s * 32 + k0a;
            float xa0 = (mrow < N_) ? kB[mrow * KS_ + da]     : 0.0f;
            float xa1 = (mrow < N_) ? kB[mrow * KS_ + da + 1] : 0.0f;
            ka[2 * v]     = (_Float16)xa0;
            ka[2 * v + 1] = (_Float16)xa1;
            int k0b = ((v < 4) ? 2 * v : 16 + 2 * (v - 4)) + 8 * half;
            int db  = s * 32 + k0b;
            // B[K=db][N=j] = query[j][db]
            qb[2 * v]     = (_Float16)q_lds[mrow][db];
            qb[2 * v + 1] = (_Float16)q_lds[mrow][db + 1];
        }
        atile = __builtin_amdgcn_wmma_f32_16x16x32_f16(
            false, ka, false, qb, (short)0, atile, false, false);
    }
#pragma unroll
    for (int r = 0; r < 8; ++r)
        attn_lds[r + 8 * half][mrow] = atile[r];
    __syncthreads();

    // ---- Step 3: per-column softmax over rows r != j, zero diagonal ----
    if (lane < N_) {
        const int j = lane;
        float m = -1e30f;
#pragma unroll
        for (int r = 0; r < N_; ++r)
            if (r != j) m = fmaxf(m, attn_lds[r][j]);
        float ex[N_];
        float sum = 0.0f;
#pragma unroll
        for (int r = 0; r < N_; ++r) {
            if (r != j) { ex[r] = __expf(attn_lds[r][j] - m); sum += ex[r]; }
            else          ex[r] = 0.0f;
        }
        float inv = 1.0f / sum;
#pragma unroll
        for (int r = 0; r < N_; ++r)
            app_out[(size_t)b * N_ * N_ + r * N_ + j] = ex[r] * inv;
    }
}

// ---------------------------------------------------------------------------
// Kernel 2: out[b,q,e] = sum_k app[b,k,q] * v[b,k,e].  Pure HBM streaming:
// 671 MB in + 671 MB out, AI ~2.5 F/B -> bandwidth bound (~58 us at 23.3 TB/s).
// float4 (B128) accesses, non-temporal hints (both streams >> 192 MB L2).
// ---------------------------------------------------------------------------
__global__ __launch_bounds__(256) void spread_kernel(
    const float* __restrict__ v, const float* __restrict__ app,
    float* __restrict__ out)
{
    __shared__ float s_app[N_ * N_];
    const int b   = blockIdx.y;
    const int tid = threadIdx.x;
    if (tid < N_ * N_) s_app[tid] = app[b * N_ * N_ + tid];
    __syncthreads();

    const size_t E4 = CHW_ / 4;
    const size_t e4 = (size_t)blockIdx.x * blockDim.x + tid;
    const f4* vb = reinterpret_cast<const f4*>(v   + (size_t)b * N_ * CHW_);
    f4*       ob = reinterpret_cast<f4*>(      out + (size_t)b * N_ * CHW_);

    f4 vv[N_];
#pragma unroll
    for (int kk = 0; kk < N_; ++kk)
        vv[kk] = __builtin_nontemporal_load(&vb[(size_t)kk * E4 + e4]);

#pragma unroll
    for (int q = 0; q < N_; ++q) {
        f4 acc = (f4)0.0f;
#pragma unroll
        for (int kk = 0; kk < N_; ++kk) {
            float a = s_app[kk * N_ + q];
            acc.x = fmaf(a, vv[kk].x, acc.x);
            acc.y = fmaf(a, vv[kk].y, acc.y);
            acc.z = fmaf(a, vv[kk].z, acc.z);
            acc.w = fmaf(a, vv[kk].w, acc.w);
        }
        __builtin_nontemporal_store(acc, &ob[(size_t)q * E4 + e4]);
    }
}

extern "C" void kernel_launch(void* const* d_in, const int* in_sizes, int n_in,
                              void* d_out, int out_size, void* d_ws, size_t ws_size,
                              hipStream_t stream) {
    const float* qu   = (const float*)d_in[0];
    const float* kmat = (const float*)d_in[1];
    const float* v    = (const float*)d_in[2];
    const float* W    = (const float*)d_in[3];
    const float* bias = (const float*)d_in[4];
    float* out = (float*)d_out;
    float* app = out + OUT_ELEMS;   // tuple output: (out, app) concatenated flat

    attn_app_kernel<<<B_, 32, 0, stream>>>(qu, kmat, W, bias, app);

    dim3 grid((unsigned)(CHW_ / 4 / 256), B_);   // 512 x 32 blocks
    spread_kernel<<<grid, 256, 0, stream>>>(v, app, out);
}
